// GriffinRecurrentBlock_31464930410573
// MI455X (gfx1250) — matrix operand order repaired and checked by reading, not compiled, not run on hardware
//
#include <hip/hip_runtime.h>
#include <hip/hip_bf16.h>
#include <math.h>
#include <cstdint>

// Problem constants (from reference)
#define DIM_D 1024
#define DIM_H 1024
#define T_LEN 4096
#define BATCH 8
#define TWIN  4
#define C_GATE 8.0f
#define M_ROWS (BATCH * T_LEN)   /* 32768 */
#define NCHUNK 16
#define CHUNK  (T_LEN / NCHUNK)  /* 256 */

typedef __bf16 bf16_t;
typedef __attribute__((ext_vector_type(16))) bf16_t v16bf;
typedef __attribute__((ext_vector_type(8)))  bf16_t v8bf;
typedef __attribute__((ext_vector_type(8)))  float  v8f;
typedef __attribute__((ext_vector_type(4)))  int    v4i;

#define AS1 __attribute__((address_space(1)))
#define AS3 __attribute__((address_space(3)))

union BFrag { v16bf v; v8bf h[2]; };

#if defined(__AMDGCN__) && __has_builtin(__builtin_amdgcn_global_load_async_to_lds_b128)
#define HAVE_ASYNC_LDS 1
#else
#define HAVE_ASYNC_LDS 0
#endif

// Copy 16 bf16 (32 bytes) global -> LDS. Async (ASYNCcnt-tracked) when available.
__device__ __forceinline__ void g2l_copy32B(const bf16_t* __restrict__ g, bf16_t* l) {
#if HAVE_ASYNC_LDS
    AS1 v4i* gp = (AS1 v4i*)(uintptr_t)g;
    AS3 v4i* lp = (AS3 v4i*)(AS3 void*)l;
    __builtin_amdgcn_global_load_async_to_lds_b128(gp, lp, 0, 0);
    __builtin_amdgcn_global_load_async_to_lds_b128(gp + 1, lp + 1, 0, 0);
#else
    const uint4* gs = (const uint4*)g;
    uint4 t0 = gs[0];
    uint4 t1 = gs[1];
    ((uint4*)l)[0] = t0;
    ((uint4*)l)[1] = t1;
#endif
}

__device__ __forceinline__ void wait_async_lds() {
#if HAVE_ASYNC_LDS
#if __has_builtin(__builtin_amdgcn_s_wait_asynccnt)
    __builtin_amdgcn_s_wait_asynccnt(0);
#else
    asm volatile("s_wait_asynccnt 0x0" ::: "memory");
#endif
#endif
}

// ---------------------------------------------------------------------------
// fp32 -> bf16 conversion (activations)
__global__ void k_cvt_bf16(const float* __restrict__ src, bf16_t* __restrict__ dst, int n) {
    int i = blockIdx.x * blockDim.x + threadIdx.x;
    if (i < n) dst[i] = (bf16_t)src[i];
}

// Pack a (K=1024, N=1024) row-major fp32 weight into bf16 B-fragment layout:
// wp[(kb/32)*(H*32) + n*32 + kk] = w[(kb+kk)*H + n]
__global__ void k_pack_w(const float* __restrict__ w, bf16_t* __restrict__ wp) {
    int i = blockIdx.x * blockDim.x + threadIdx.x;   // over 1024*1024
    int kb  = i >> 15;          // / (H*32)
    int rem = i & 32767;
    int n   = rem >> 5;
    int kk  = rem & 31;
    wp[i] = (bf16_t)w[(size_t)(kb * 32 + kk) * DIM_H + n];
}

// ---------------------------------------------------------------------------
// WMMA GEMM:  C(M=32768, N=1024) = A(M,1024) * W(1024,1024), fused epilogues.
// 256 threads = 8 waves (4 along M, 2 along N). Wave tile 32x64 (8 WMMA accs).
// Block tile 128x128. B tile staged in LDS via async global->LDS, double buffered.
// MODE 0: out0 = acc + bias                        (x1)
// MODE 1: out0 = gelu_exact(acc + bias)            (y)
// MODE 2: gate_a path: out0 = a, out1 = multiplier (aux0 = a_param)
// MODE 3: gate_x path: out0 = sigmoid(acc+bias) * aux0(x2) * out0(multiplier)
template <int MODE>
__global__ __launch_bounds__(256) void k_gemm(
    const bf16_t* __restrict__ A, const bf16_t* __restrict__ Wp,
    const float* __restrict__ bias,
    float* __restrict__ out0, float* __restrict__ out1,
    const float* __restrict__ aux0)
{
    __shared__ bf16_t sB[2][128 * 32];   // 2 x 8KB B tiles (128 cols x 32 k)

    const int lane = threadIdx.x & 31;
    const int wave = threadIdx.x >> 5;
    const int wm   = wave & 3;       // 4 waves along M
    const int wn   = wave >> 2;      // 2 waves along N
    const int half = lane >> 4;
    const int mr   = lane & 15;

    const int rowb  = blockIdx.x * 128 + wm * 32 + mr;   // first A-frag row
    const int nbase = blockIdx.y * 128 + wn * 64;        // wave's first column

    const bf16_t* ar0 = A + (size_t)rowb * DIM_D;        // rows rowb, rowb+16
    const bf16_t* ar1 = ar0 + (size_t)16 * DIM_D;

    // per-thread B staging source/dest (32 bytes each, whole block covers 8KB)
    const bf16_t* wsrc = Wp + ((size_t)blockIdx.y * 128) * 32 + (size_t)threadIdx.x * 16;
    const int sOff = threadIdx.x * 16;

    v8f acc[8] = {};

    constexpr int NKB = DIM_D / 32;   // 32 k-steps
    g2l_copy32B(wsrc, &sB[0][sOff]);  // preload k-step 0

    for (int kbi = 0; kbi < NKB; ++kbi) {
        wait_async_lds();
        __syncthreads();
        if (kbi + 1 < NKB)
            g2l_copy32B(wsrc + (size_t)(kbi + 1) * (DIM_H * 32), &sB[(kbi + 1) & 1][sOff]);

        const int kb = kbi * 32;
        // A fragments (16x32 bf16): lanes0-15 K=kb+0..7|16..23, lanes16-31 K=kb+8..15|24..31
        BFrag a0, a1;
        a0.h[0] = *(const v8bf*)(ar0 + kb + half * 8);
        a0.h[1] = *(const v8bf*)(ar0 + kb + 16 + half * 8);
        a1.h[0] = *(const v8bf*)(ar1 + kb + half * 8);
        a1.h[1] = *(const v8bf*)(ar1 + kb + 16 + half * 8);

        const bf16_t* bbase = &sB[kbi & 1][(wn * 64) * 32];
#pragma unroll
        for (int s = 0; s < 4; ++s) {
            // B fragment (32x16 bf16) from LDS: col = mr, lanes0-15 K=0..7|16..23, lanes16-31 K=8..15|24..31
            const bf16_t* bp = bbase + (s * 16 + mr) * 32 + half * 8;
            BFrag bf;
            bf.h[0] = *(const v8bf*)(bp);
            bf.h[1] = *(const v8bf*)(bp + 16);
            acc[s] = __builtin_amdgcn_wmma_f32_16x16x32_bf16(
                false, a0.v, false, bf.v, (short)0, acc[s], false, false);
            acc[4 + s] = __builtin_amdgcn_wmma_f32_16x16x32_bf16(
                false, a1.v, false, bf.v, (short)0, acc[4 + s], false, false);
        }
    }

    // Epilogue: C/D layout: VGPR r -> row half*8+r, col = mr
#pragma unroll
    for (int mi = 0; mi < 2; ++mi) {
        const int rbase = blockIdx.x * 128 + wm * 32 + mi * 16 + half * 8;
#pragma unroll
        for (int r = 0; r < 8; ++r) {
            const int orow = rbase + r;
            const int t = orow & (T_LEN - 1);
            const size_t rowoff = (size_t)orow * DIM_H;
#pragma unroll
            for (int s = 0; s < 4; ++s) {
                const int col = nbase + s * 16 + mr;
                const size_t idx = rowoff + col;
                float v = acc[mi * 4 + s][r] + bias[col];
                if (MODE == 0) {
                    out0[idx] = v;
                } else if (MODE == 1) {
                    out0[idx] = 0.5f * v * (1.0f + erff(v * 0.70710678118654752f));
                } else if (MODE == 2) {
                    float ti = 1.0f / (1.0f + expf(-v));              // sigmoid(gate_a)
                    float ap = aux0[col];
                    float sp = (ap > 20.0f) ? ap : log1pf(expf(ap));  // softplus(a_param)
                    float la = -C_GATE * ti * sp;                     // log_a
                    float a, m;
                    if (t == 0) { a = 0.0f; m = 1.0f; }
                    else {
                        a = expf(la);
                        float e2 = expf(2.0f * la);
                        m = sqrtf(fmaxf(1.0f - e2, 0.0f));
                    }
                    out0[idx] = a;
                    out1[idx] = m;
                } else { // MODE 3
                    float gx = 1.0f / (1.0f + expf(-v));              // sigmoid(gate_x)
                    out0[idx] = gx * aux0[idx] * out0[idx];           // gx * x2 * multiplier
                }
            }
        }
    }
}

// ---------------------------------------------------------------------------
// Depthwise causal conv over T (window 4): x2 = sum_i cw[i]*x1[t-3+i] + cb
__global__ void k_conv(const float* __restrict__ x1, const float* __restrict__ cw,
                       const float* __restrict__ cb, float* __restrict__ x2f,
                       bf16_t* __restrict__ x2b)
{
    int i = blockIdx.x * blockDim.x + threadIdx.x;   // over B*T*H
    int h  = i & (DIM_H - 1);
    int bt = i >> 10;
    int t  = bt & (T_LEN - 1);
    float acc = cb[h];
#pragma unroll
    for (int j = 0; j < TWIN; ++j) {
        int tt = t - (TWIN - 1) + j;
        if (tt >= 0)
            acc += cw[j * DIM_H + h] * x1[(size_t)i + (size_t)(j - (TWIN - 1)) * DIM_H];
    }
    x2f[i] = acc;
    x2b[i] = (bf16_t)acc;
}

// ---------------------------------------------------------------------------
// Chunked linear-recurrence scan:  h[t] = a[t]*h[t-1] + nx[t]
// Pass 1: per-(b,chunk,h) local scan; av overwritten with in-chunk prefix product.
__global__ void k_scan1(float* __restrict__ av, const float* __restrict__ nx,
                        float* __restrict__ hloc,
                        float* __restrict__ cA, float* __restrict__ cH)
{
    int tid = blockIdx.x * blockDim.x + threadIdx.x;  // B*NCHUNK*H
    int h   = tid & (DIM_H - 1);
    int rem = tid >> 10;
    int c   = rem & (NCHUNK - 1);
    int b   = rem >> 4;
    size_t base = ((size_t)b * T_LEN + (size_t)c * CHUNK) * DIM_H + h;
    float hp = 0.0f, P = 1.0f;
    for (int t0 = 0; t0 < CHUNK; ++t0) {
        size_t idx = base + (size_t)t0 * DIM_H;
        float a = av[idx];
        float x = nx[idx];
        hp = fmaf(a, hp, x);
        P *= a;
        av[idx]   = P;    // in-chunk prefix product (incl. current a)
        hloc[idx] = hp;   // local h (zero initial state)
    }
    int ch = (b * DIM_H + h) * NCHUNK + c;
    cA[ch] = P;
    cH[ch] = hp;
}

// Pass 2: sequential carry across chunks per channel (16 steps, tiny).
__global__ void k_scan2(const float* __restrict__ cA, const float* __restrict__ cH,
                        float* __restrict__ carryIn)
{
    int tid = blockIdx.x * blockDim.x + threadIdx.x;  // B*H
    float cur = 0.0f;
    for (int c = 0; c < NCHUNK; ++c) {
        int ix = tid * NCHUNK + c;
        carryIn[ix] = cur;
        cur = fmaf(cA[ix], cur, cH[ix]);
    }
}

// Pass 3: h = hloc + prefixA*carry ; fuse out = h*y, emit h and hn.
__global__ void k_scan3(const float* __restrict__ av, const float* __restrict__ carryIn,
                        const float* __restrict__ yv, float* __restrict__ out,
                        float* __restrict__ hout, float* __restrict__ hn)
{
    size_t i = (size_t)blockIdx.x * blockDim.x + threadIdx.x;  // B*T*H
    int h  = (int)(i & (DIM_H - 1));
    size_t bt = i >> 10;
    int t  = (int)(bt & (T_LEN - 1));
    int b  = (int)(bt >> 12);
    int c  = t / CHUNK;
    float carry = carryIn[(size_t)(b * DIM_H + h) * NCHUNK + c];
    float hf = fmaf(av[i], carry, hout[i]);
    hout[i] = hf;
    out[i]  = hf * yv[i];
    if (t == T_LEN - 1) hn[(size_t)b * DIM_H + h] = hf;
}

// ---------------------------------------------------------------------------
extern "C" void kernel_launch(void* const* d_in, const int* in_sizes, int n_in,
                              void* d_out, int out_size, void* d_ws, size_t ws_size,
                              hipStream_t stream)
{
    const float* x    = (const float*)d_in[0];
    const float* w_x  = (const float*)d_in[1];
    const float* b_x  = (const float*)d_in[2];
    const float* w_y  = (const float*)d_in[3];
    const float* b_y  = (const float*)d_in[4];
    const float* cw   = (const float*)d_in[5];
    const float* cb   = (const float*)d_in[6];
    const float* gi_w = (const float*)d_in[7];
    const float* gi_b = (const float*)d_in[8];
    const float* ga_w = (const float*)d_in[9];
    const float* ga_b = (const float*)d_in[10];
    const float* a_p  = (const float*)d_in[11];

    const size_t NE = (size_t)M_ROWS * DIM_H;       // 33,554,432 elems
    const size_t WE = (size_t)DIM_D * DIM_H;        // 1,048,576 elems

    char* p = (char*)d_ws;
    bf16_t* xb  = (bf16_t*)p; p += NE * sizeof(bf16_t);
    bf16_t* wxp = (bf16_t*)p; p += WE * sizeof(bf16_t);
    bf16_t* wyp = (bf16_t*)p; p += WE * sizeof(bf16_t);
    bf16_t* gip = (bf16_t*)p; p += WE * sizeof(bf16_t);
    bf16_t* gap = (bf16_t*)p; p += WE * sizeof(bf16_t);
    float*  x1  = (float*)p;  p += NE * sizeof(float);
    float*  x2f = (float*)p;  p += NE * sizeof(float);
    bf16_t* x2b = (bf16_t*)p; p += NE * sizeof(bf16_t);
    float*  yv  = (float*)p;  p += NE * sizeof(float);
    float*  av  = (float*)p;  p += NE * sizeof(float);
    float*  nxv = (float*)p;  p += NE * sizeof(float);
    const size_t CE = (size_t)BATCH * DIM_H * NCHUNK;  // 131,072
    float* cA  = (float*)p; p += CE * sizeof(float);
    float* cH  = (float*)p; p += CE * sizeof(float);
    float* cin = (float*)p; p += CE * sizeof(float);

    float* out  = (float*)d_out;
    float* hout = out + NE;
    float* hn   = out + 2 * NE;

    dim3 blk(256);

    // 1) convert x to bf16; pack all 4 weight matrices to bf16 fragment layout
    k_cvt_bf16<<<dim3((unsigned)(NE / 256)), blk, 0, stream>>>(x, xb, (int)NE);
    k_pack_w<<<dim3((unsigned)(WE / 256)), blk, 0, stream>>>(w_x,  wxp);
    k_pack_w<<<dim3((unsigned)(WE / 256)), blk, 0, stream>>>(w_y,  wyp);
    k_pack_w<<<dim3((unsigned)(WE / 256)), blk, 0, stream>>>(gi_w, gip);
    k_pack_w<<<dim3((unsigned)(WE / 256)), blk, 0, stream>>>(ga_w, gap);

    dim3 gg(M_ROWS / 128, DIM_H / 128);   // (256, 8)

    // 2) x1 = x@w_x + b_x ;  y = gelu(x@w_y + b_y)
    k_gemm<0><<<gg, blk, 0, stream>>>(xb, wxp, b_x, x1, nullptr, nullptr);
    k_gemm<1><<<gg, blk, 0, stream>>>(xb, wyp, b_y, yv, nullptr, nullptr);

    // 3) depthwise causal conv -> x2 (fp32 + bf16)
    k_conv<<<dim3((unsigned)(NE / 256)), blk, 0, stream>>>(x1, cw, cb, x2f, x2b);

    // 4) gate_a path: a -> av, multiplier -> nxv
    k_gemm<2><<<gg, blk, 0, stream>>>(x2b, gap, ga_b, av, nxv, a_p);
    // 5) gate_x path: nxv = sigmoid(x2@gi_w+gi_b) * x2 * multiplier
    k_gemm<3><<<gg, blk, 0, stream>>>(x2b, gip, gi_b, nxv, nullptr, x2f);

    // 6) chunked scan + fused output
    k_scan1<<<dim3((unsigned)((BATCH * NCHUNK * DIM_H) / 256)), blk, 0, stream>>>(av, nxv, hout, cA, cH);
    k_scan2<<<dim3((unsigned)((BATCH * DIM_H) / 256)), blk, 0, stream>>>(cA, cH, cin);
    k_scan3<<<dim3((unsigned)(NE / 256)), blk, 0, stream>>>(av, cin, yv, out, hout, hn);
}
